// SpatialRelationLayer_64287070486886
// MI455X (gfx1250) — matrix operand order repaired and checked by reading, not compiled, not run on hardware
//
#include <hip/hip_runtime.h>
#include <math.h>

typedef __attribute__((ext_vector_type(16))) _Float16 v16h;
typedef __attribute__((ext_vector_type(8)))  float    v8f;
typedef __attribute__((ext_vector_type(4)))  unsigned u32x4;
typedef __attribute__((ext_vector_type(2)))  unsigned u32x2;

union Frag16 {
  v16h  v;
  unsigned u[8];
  u32x4 q[2];
};

union H4pack {
  _Float16 h[4];
  u32x2 u;
};

__device__ __forceinline__ v8f vzero() { v8f z = {}; return z; }

// ---------------------------------------------------------------------------
// A-fragment load from LDS (row-major f16 tile, KP halves per row).
// 16-bit A 16x32 layout: lanes 0-15 = rows m; halves h -> K = (h<8?h:8+h) + (lane>>4)*8
__device__ __forceinline__ v16h load_a_frag(const unsigned* xl, int KP, int kt, int lane) {
  int m  = lane & 15;
  int hi = lane >> 4;
  int b32 = (m * KP + kt * 32 + hi * 8) >> 1;   // u32 index
  Frag16 f;
#pragma unroll
  for (int v = 0; v < 4; ++v) f.u[v] = xl[b32 + v];
#pragma unroll
  for (int v = 0; v < 4; ++v) f.u[4 + v] = xl[b32 + 8 + v];
  return f.v;
}

// B-fragment: weights pre-packed so each lane reads 16 contiguous halves (2xB128).
__device__ __forceinline__ v16h load_b_frag(const _Float16* wt, int lane) {
  Frag16 f;
  const u32x4* q = (const u32x4*)(wt + lane * 16);
  f.q[0] = q[0];
  f.q[1] = q[1];
  return f.v;
}

// ---------------------------------------------------------------------------
// Fused MLP for MT 16-row tiles (MT*16 rows) held in LDS.
// Hidden padded 132 -> 160 (10 n-tiles, 5 k-tiles). B fragments reused across
// MT m-tiles; A fragments held in registers across a block of 5 n-tiles.
template <int KP0, int MT>
__device__ __forceinline__ void mlp_run(
    const unsigned* xl, unsigned* h0, unsigned* h1,
    const _Float16* __restrict__ W0p, const float* __restrict__ b0,
    const _Float16* __restrict__ W1p, const float* __restrict__ b1,
    const _Float16* __restrict__ W2p, const float* __restrict__ b2,
    float* __restrict__ out, long long rowBase, int lane) {
  constexpr int KT0 = KP0 / 32;
  const int laneLo = lane & 15;
  const int laneHi = lane >> 4;

  // ---- Layer 0: X[MT*16 x KP0] @ W0[KP0 x 160] -> H0 (relu, f16) ----
  for (int ntb = 0; ntb < 10; ntb += 5) {
    v8f acc[5][MT];
#pragma unroll
    for (int n5 = 0; n5 < 5; ++n5)
#pragma unroll
      for (int mt = 0; mt < MT; ++mt) acc[n5][mt] = vzero();
    for (int kt = 0; kt < KT0; ++kt) {
      v16h a[MT];
#pragma unroll
      for (int mt = 0; mt < MT; ++mt)
        a[mt] = load_a_frag(xl + mt * 8 * KP0, KP0, kt, lane);
#pragma unroll
      for (int n5 = 0; n5 < 5; ++n5) {
        v16h b = load_b_frag(W0p + (size_t)(kt * 10 + ntb + n5) * 512, lane);
#pragma unroll
        for (int mt = 0; mt < MT; ++mt)
          acc[n5][mt] = __builtin_amdgcn_wmma_f32_16x16x32_f16(
              false, a[mt], false, b, (short)0, acc[n5][mt], false, false);
      }
    }
    _Float16* hh = (_Float16*)h0;
#pragma unroll
    for (int n5 = 0; n5 < 5; ++n5) {
      int n = (ntb + n5) * 16 + laneLo;
      float bias = (n < 132) ? b0[n] : 0.0f;
#pragma unroll
      for (int mt = 0; mt < MT; ++mt)
#pragma unroll
        for (int r = 0; r < 8; ++r) {
          float h = acc[n5][mt][r] + bias;
          h = h > 0.0f ? h : 0.0f;
          hh[(mt * 16 + laneHi * 8 + r) * 160 + n] = (_Float16)h;
        }
    }
  }
  __syncthreads();

  // ---- Layer 1: H0 @ W1[160 x 160] -> H1 (relu, f16; H1 aliases X) ----
  for (int ntb = 0; ntb < 10; ntb += 5) {
    v8f acc[5][MT];
#pragma unroll
    for (int n5 = 0; n5 < 5; ++n5)
#pragma unroll
      for (int mt = 0; mt < MT; ++mt) acc[n5][mt] = vzero();
    for (int kt = 0; kt < 5; ++kt) {
      v16h a[MT];
#pragma unroll
      for (int mt = 0; mt < MT; ++mt)
        a[mt] = load_a_frag(h0 + mt * 8 * 160, 160, kt, lane);
#pragma unroll
      for (int n5 = 0; n5 < 5; ++n5) {
        v16h b = load_b_frag(W1p + (size_t)(kt * 10 + ntb + n5) * 512, lane);
#pragma unroll
        for (int mt = 0; mt < MT; ++mt)
          acc[n5][mt] = __builtin_amdgcn_wmma_f32_16x16x32_f16(
              false, a[mt], false, b, (short)0, acc[n5][mt], false, false);
      }
    }
    _Float16* hh = (_Float16*)h1;
#pragma unroll
    for (int n5 = 0; n5 < 5; ++n5) {
      int n = (ntb + n5) * 16 + laneLo;
      float bias = (n < 132) ? b1[n] : 0.0f;
#pragma unroll
      for (int mt = 0; mt < MT; ++mt)
#pragma unroll
        for (int r = 0; r < 8; ++r) {
          float h = acc[n5][mt][r] + bias;
          h = h > 0.0f ? h : 0.0f;
          hh[(mt * 16 + laneHi * 8 + r) * 160 + n] = (_Float16)h;
        }
    }
  }
  __syncthreads();

  // ---- Layer 2: H1 @ W2[160 x 16] -> sigmoid(5*(.+b2)) -> global f32 ----
  {
    v8f acc[MT];
#pragma unroll
    for (int mt = 0; mt < MT; ++mt) acc[mt] = vzero();
    for (int kt = 0; kt < 5; ++kt) {
      v16h b = load_b_frag(W2p + (size_t)kt * 512, lane);
#pragma unroll
      for (int mt = 0; mt < MT; ++mt) {
        v16h a = load_a_frag(h1 + mt * 8 * 160, 160, kt, lane);
        acc[mt] = __builtin_amdgcn_wmma_f32_16x16x32_f16(
            false, a, false, b, (short)0, acc[mt], false, false);
      }
    }
    int n = laneLo;
    float bias = b2[n];
#pragma unroll
    for (int mt = 0; mt < MT; ++mt)
#pragma unroll
      for (int r = 0; r < 8; ++r) {
        long long row = rowBase + mt * 16 + laneHi * 8 + r;
        float y = acc[mt][r] + bias;
        y = 1.0f / (1.0f + __expf(-5.0f * y));
        out[row * 16 + n] = y;
      }
  }
}

// ---------------------------------------------------------------------------
// Weight packing: W[din x dout] f32 -> f16 tiles in B-fragment lane order.
//   k = kt*32 + (lane>>4)*8 + (h&7) + (h>>3)*16 ; n = nt*16 + (lane&15); zero pad OOB.
__global__ void pack_w_kernel(const float* __restrict__ W, int din, int dout, int NT,
                              _Float16* __restrict__ out) {
  int tile = blockIdx.x;
  int kt = tile / NT;
  int nt = tile - kt * NT;
  for (int e = threadIdx.x; e < 512; e += blockDim.x) {
    int lane = e >> 4, h = e & 15;
    int k = kt * 32 + (lane >> 4) * 8 + (h & 7) + (h >> 3) * 16;
    int n = nt * 16 + (lane & 15);
    float v = (k < din && n < dout) ? W[(size_t)k * dout + n] : 0.0f;
    out[(size_t)tile * 512 + e] = (_Float16)v;
  }
}

// max over N of Q: out[bt*64+d] = max_n Q[bt,n,d]   (32*64 outputs)
__global__ void qmax_kernel(const float* __restrict__ Q, float* __restrict__ out) {
  int idx = blockIdx.x * blockDim.x + threadIdx.x;
  if (idx >= 32 * 64) return;
  int bt = idx >> 6, d = idx & 63;
  float m = -3.4e38f;
  for (int n = 0; n < 128; ++n)
    m = fmaxf(m, Q[((size_t)bt * 128 + n) * 64 + d]);
  out[idx] = m;
}

// R reductions: red0[bt,n,d] = max_i R[bt,i,n,d]; red1[bt,n,d] = max_j R[bt,n,j,d]
__global__ void rred_kernel(const float* __restrict__ R,
                            float* __restrict__ out0, float* __restrict__ out1) {
  int idx = blockIdx.x * blockDim.x + threadIdx.x;
  if (idx >= 4096 * 32) return;
  int d = idx & 31, n = (idx >> 5) & 127, bt = idx >> 12;
  size_t base = (size_t)bt * 128 * 128 * 32;
  float m0 = -3.4e38f, m1 = -3.4e38f;
  for (int k = 0; k < 128; ++k) {
    m0 = fmaxf(m0, R[base + ((size_t)k * 128 + n) * 32 + d]);
    m1 = fmaxf(m1, R[base + ((size_t)n * 128 + k) * 32 + d]);
  }
  out0[idx] = m0;
  out1[idx] = m1;
}

// ---------------------------------------------------------------------------
// R path: rows = 524288; X = concat(R[g], Q[bt,j], Q[bt,i]) -> 160. MT=4.
__global__ __launch_bounds__(32) void r_path_kernel(
    const float* __restrict__ R, const float* __restrict__ Q,
    const _Float16* W0p, const float* b0, const _Float16* W1p, const float* b1,
    const _Float16* W2p, const float* b2, float* __restrict__ out) {
  constexpr int KP0 = 160, MT = 4;
  constexpr int XBUF = MT * 16 * 160 / 2;  // u32 words (X aliases H1)
  constexpr int HBUF = MT * 16 * 160 / 2;
  __shared__ unsigned smem[XBUF + HBUF];
  unsigned* xl = smem;
  unsigned* h1 = smem;          // aliases X (X dead after layer 0)
  unsigned* h0 = smem + XBUF;
  int lane = threadIdx.x & 31;

  long long g0 = (long long)blockIdx.x * (MT * 16);
  int bt = (int)(g0 >> 14);
  int i  = (int)((g0 >> 7) & 127);
  int j0 = (int)(g0 & 127);

  _Float16* xh = (_Float16*)xl;
  for (int e = lane; e < MT * 16 * (KP0 / 4); e += 32) {
    int m = e / (KP0 / 4);
    int f4 = (e - m * (KP0 / 4)) * 4;
    const float* src;
    if (f4 < 32)      src = R + (size_t)(g0 + m) * 32 + f4;
    else if (f4 < 96) src = Q + ((size_t)bt * 128 + j0 + m) * 64 + (f4 - 32);
    else              src = Q + ((size_t)bt * 128 + i) * 64 + (f4 - 96);
    float4 v = *(const float4*)src;
    H4pack t;
    t.h[0] = (_Float16)v.x; t.h[1] = (_Float16)v.y;
    t.h[2] = (_Float16)v.z; t.h[3] = (_Float16)v.w;
    *(u32x2*)(xh + (size_t)m * KP0 + f4) = t.u;
  }
  __syncthreads();
  mlp_run<KP0, MT>(xl, h0, h1, W0p, b0, W1p, b1, W2p, b2, out, g0, lane);
}

// Q path: rows = 4096; X = concat(P[bt], Q[g], red0[g], red1[g]) -> 192. MT=4.
__global__ __launch_bounds__(32) void q_path_kernel(
    const float* __restrict__ P, const float* __restrict__ Q,
    const float* __restrict__ red0, const float* __restrict__ red1,
    const _Float16* W0p, const float* b0, const _Float16* W1p, const float* b1,
    const _Float16* W2p, const float* b2, float* __restrict__ out) {
  constexpr int KP0 = 192, MT = 4;
  constexpr int XBUF = MT * 16 * 192 / 2;  // X needs 192; H1 (160-wide) aliases it
  constexpr int HBUF = MT * 16 * 160 / 2;
  __shared__ unsigned smem[XBUF + HBUF];
  unsigned* xl = smem;
  unsigned* h1 = smem;
  unsigned* h0 = smem + XBUF;
  int lane = threadIdx.x & 31;

  long long g0 = (long long)blockIdx.x * (MT * 16);
  int bt = (int)(g0 >> 7);

  _Float16* xh = (_Float16*)xl;
  for (int e = lane; e < MT * 16 * (KP0 / 4); e += 32) {
    int m = e / (KP0 / 4);
    int f4 = (e - m * (KP0 / 4)) * 4;
    const float* src;
    if (f4 < 64)       src = P + (size_t)bt * 64 + f4;
    else if (f4 < 128) src = Q + (size_t)(g0 + m) * 64 + (f4 - 64);
    else if (f4 < 160) src = red0 + (size_t)(g0 + m) * 32 + (f4 - 128);
    else               src = red1 + (size_t)(g0 + m) * 32 + (f4 - 160);
    float4 v = *(const float4*)src;
    H4pack t;
    t.h[0] = (_Float16)v.x; t.h[1] = (_Float16)v.y;
    t.h[2] = (_Float16)v.z; t.h[3] = (_Float16)v.w;
    *(u32x2*)(xh + (size_t)m * KP0 + f4) = t.u;
  }
  __syncthreads();
  mlp_run<KP0, MT>(xl, h0, h1, W0p, b0, W1p, b1, W2p, b2, out, g0, lane);
}

// P path: rows = 32; X = concat(maxQ[g], P[g]) -> 128. MT=2, one block.
__global__ __launch_bounds__(32) void p_path_kernel(
    const float* __restrict__ P, const float* __restrict__ qmax,
    const _Float16* W0p, const float* b0, const _Float16* W1p, const float* b1,
    const _Float16* W2p, const float* b2, float* __restrict__ out) {
  constexpr int KP0 = 128, MT = 2;
  constexpr int XBUF = MT * 16 * 160 / 2;  // sized for H1 alias (160-wide)
  constexpr int HBUF = MT * 16 * 160 / 2;
  __shared__ unsigned smem[XBUF + HBUF];
  unsigned* xl = smem;
  unsigned* h1 = smem;
  unsigned* h0 = smem + XBUF;
  int lane = threadIdx.x & 31;

  long long g0 = 0;

  _Float16* xh = (_Float16*)xl;
  for (int e = lane; e < MT * 16 * (KP0 / 4); e += 32) {
    int m = e / (KP0 / 4);
    int f4 = (e - m * (KP0 / 4)) * 4;
    const float* src;
    if (f4 < 64) src = qmax + (size_t)m * 64 + f4;
    else         src = P + (size_t)m * 64 + (f4 - 64);
    float4 v = *(const float4*)src;
    H4pack t;
    t.h[0] = (_Float16)v.x; t.h[1] = (_Float16)v.y;
    t.h[2] = (_Float16)v.z; t.h[3] = (_Float16)v.w;
    *(u32x2*)(xh + (size_t)m * KP0 + f4) = t.u;
  }
  __syncthreads();
  mlp_run<KP0, MT>(xl, h0, h1, W0p, b0, W1p, b1, W2p, b2, out, g0, lane);
}

// ---------------------------------------------------------------------------
extern "C" void kernel_launch(void* const* d_in, const int* in_sizes, int n_in,
                              void* d_out, int out_size, void* d_ws, size_t ws_size,
                              hipStream_t stream) {
  const float* P = (const float*)d_in[0];
  const float* Q = (const float*)d_in[1];
  const float* R = (const float*)d_in[2];
  const float* PW0 = (const float*)d_in[3];  const float* Pb0 = (const float*)d_in[4];
  const float* PW1 = (const float*)d_in[5];  const float* Pb1 = (const float*)d_in[6];
  const float* PW2 = (const float*)d_in[7];  const float* Pb2 = (const float*)d_in[8];
  const float* QW0 = (const float*)d_in[9];  const float* Qb0 = (const float*)d_in[10];
  const float* QW1 = (const float*)d_in[11]; const float* Qb1 = (const float*)d_in[12];
  const float* QW2 = (const float*)d_in[13]; const float* Qb2 = (const float*)d_in[14];
  const float* RW0 = (const float*)d_in[15]; const float* Rb0 = (const float*)d_in[16];
  const float* RW1 = (const float*)d_in[17]; const float* Rb1 = (const float*)d_in[18];
  const float* RW2 = (const float*)d_in[19]; const float* Rb2 = (const float*)d_in[20];

  char* ws = (char*)d_ws;
  size_t off = 0;
  auto take = [&](size_t bytes) -> char* {
    char* p = ws + off;
    off = (off + bytes + 1023) & ~(size_t)1023;
    return p;
  };
  _Float16* wpP0 = (_Float16*)take(4 * 10 * 512 * 2);  // din 128
  _Float16* wpP1 = (_Float16*)take(5 * 10 * 512 * 2);
  _Float16* wpP2 = (_Float16*)take(5 * 1 * 512 * 2);
  _Float16* wpQ0 = (_Float16*)take(6 * 10 * 512 * 2);  // din 192
  _Float16* wpQ1 = (_Float16*)take(5 * 10 * 512 * 2);
  _Float16* wpQ2 = (_Float16*)take(5 * 1 * 512 * 2);
  _Float16* wpR0 = (_Float16*)take(5 * 10 * 512 * 2);  // din 160
  _Float16* wpR1 = (_Float16*)take(5 * 10 * 512 * 2);
  _Float16* wpR2 = (_Float16*)take(5 * 1 * 512 * 2);
  float* qmax = (float*)take(2048 * 4);
  float* red0 = (float*)take(131072 * 4);
  float* red1 = (float*)take(131072 * 4);

  // 1) Pack weights (f16, B-fragment layout, zero-padded)
  pack_w_kernel<<<4 * 10, 256, 0, stream>>>(PW0, 128, 132, 10, wpP0);
  pack_w_kernel<<<5 * 10, 256, 0, stream>>>(PW1, 132, 132, 10, wpP1);
  pack_w_kernel<<<5 * 1,  256, 0, stream>>>(PW2, 132, 16, 1, wpP2);
  pack_w_kernel<<<6 * 10, 256, 0, stream>>>(QW0, 192, 132, 10, wpQ0);
  pack_w_kernel<<<5 * 10, 256, 0, stream>>>(QW1, 132, 132, 10, wpQ1);
  pack_w_kernel<<<5 * 1,  256, 0, stream>>>(QW2, 132, 16, 1, wpQ2);
  pack_w_kernel<<<5 * 10, 256, 0, stream>>>(RW0, 160, 132, 10, wpR0);
  pack_w_kernel<<<5 * 10, 256, 0, stream>>>(RW1, 132, 132, 10, wpR1);
  pack_w_kernel<<<5 * 1,  256, 0, stream>>>(RW2, 132, 16, 1, wpR2);

  // 2) Reductions
  qmax_kernel<<<(2048 + 255) / 256, 256, 0, stream>>>(Q, qmax);
  rred_kernel<<<(131072 + 255) / 256, 256, 0, stream>>>(R, red0, red1);

  // 3) Fused WMMA MLPs (outputs concatenated: P_out | Q_out | R_out)
  float* outP = (float*)d_out;            // 32*16   = 512
  float* outQ = outP + 512;               // 4096*16 = 65536
  float* outR = outQ + 65536;             // 524288*16

  p_path_kernel<<<1, 32, 0, stream>>>(P, qmax, wpP0, Pb0, wpP1, Pb1, wpP2, Pb2, outP);
  q_path_kernel<<<4096 / 64, 32, 0, stream>>>(
      P, Q, red0, red1, wpQ0, Qb0, wpQ1, Qb1, wpQ2, Qb2, outQ);
  r_path_kernel<<<524288 / 64, 32, 0, stream>>>(
      R, Q, wpR0, Rb0, wpR1, Rb1, wpR2, Rb2, outR);
}
